// CustomRNN_41764261986307
// MI455X (gfx1250) — matrix operand order repaired and checked by reading, not compiled, not run on hardware
//
#include <hip/hip_runtime.h>
#include <hip/hip_bf16.h>

// ---------------- problem constants ----------------
constexpr int    Bz = 128;        // batch
constexpr int    Tz = 512;        // time steps
constexpr int    Vz = 128;        // vocab (layer0 input width)
constexpr int    Hz = 1024;       // hidden
constexpr int    H4 = 4 * Hz;     // gate width
constexpr size_t BH = (size_t)Bz * Hz;

constexpr int GRID_BLOCKS   = 64;   // 512 waves = 8 batch-tiles x 64 col-tiles
constexpr int BLOCK_THREADS = 256;

// ---------------- WMMA types ----------------
typedef __attribute__((ext_vector_type(16))) __bf16 v16bf;
typedef __attribute__((ext_vector_type(8)))  float  v8f;

union ABu { v16bf v; uint4 q[2]; };

__device__ __forceinline__ float fast_sigmoid(float x) {
    return 1.0f / (1.0f + __expf(-x));
}
__device__ __forceinline__ float fast_tanh(float x) {
    return 1.0f - 2.0f / (__expf(2.0f * x) + 1.0f);   // tanh(x) = 1 - 2/(e^2x+1)
}

// Accumulate a [16 x klen] * [klen x 16] product into 4 gate accumulators.
// Abase: base of A matrix; aoff: per-lane element offset of this lane's A row.
// Wbase: base of transposed weights [4H][klen]; woff[g]: per-lane element
// offset of gate-g's B row (contiguous K). zo: opaque zero (anti-LICM).
static __device__ __forceinline__ void wmma_kloop(
    v8f acc[4],
    const __hip_bfloat16* __restrict__ Abase, int aoff,
    const __hip_bfloat16* __restrict__ Wbase, const int woff[4], int zo,
    int klen, int half)
{
    for (int kb = 0; kb < klen; kb += 32) {
        ABu a;
        const __hip_bfloat16* ap = Abase + aoff + kb + half * 8;
        a.q[0] = *(const uint4*)(ap);        // K: base+{0..7} | {8..15}
        a.q[1] = *(const uint4*)(ap + 16);   // K: base+{16..23} | {24..31}
#pragma unroll
        for (int g = 0; g < 4; ++g) {
            ABu b;
            const __hip_bfloat16* bp = Wbase + (woff[g] + zo) + kb + half * 8;
            b.q[0] = *(const uint4*)(bp);
            b.q[1] = *(const uint4*)(bp + 16);
            acc[g] = __builtin_amdgcn_wmma_f32_16x16x32_bf16(
                false, a.v, false, b.v, (short)0, acc[g], false, false);
        }
    }
}

// Device-wide sense/generation barrier. bar[0] = arrival counter,
// bar[32] = generation (separate cachelines). All 64 blocks are co-resident.
__device__ __forceinline__ void grid_barrier(unsigned* bar, unsigned* lgen)
{
    __threadfence();           // release my stores to device scope
    __syncthreads();
    if (threadIdx.x == 0) {
        const unsigned g = *lgen;
        const unsigned arrived = __hip_atomic_fetch_add(
            &bar[0], 1u, __ATOMIC_ACQ_REL, __HIP_MEMORY_SCOPE_AGENT);
        if (arrived == (unsigned)(GRID_BLOCKS - 1)) {
            __hip_atomic_store(&bar[0], 0u, __ATOMIC_RELAXED, __HIP_MEMORY_SCOPE_AGENT);
            __hip_atomic_store(&bar[32], g + 1u, __ATOMIC_RELEASE, __HIP_MEMORY_SCOPE_AGENT);
        } else {
            while (__hip_atomic_load(&bar[32], __ATOMIC_ACQUIRE,
                                     __HIP_MEMORY_SCOPE_AGENT) == g) {
                __builtin_amdgcn_s_sleep(1);
            }
        }
        *lgen = g + 1u;
    }
    __syncthreads();
    __threadfence();           // acquire: discard stale cached lines
}

// Persistent 2-layer LSTM scan. Each wave owns one 16(batch)x16(hidden) tile
// for BOTH layers across all 512 steps; c-state stays in VGPRs the whole run.
__global__ void lstm_persistent(
    const __hip_bfloat16* __restrict__ xbf,    // [B][T][V] bf16
    const __hip_bfloat16* __restrict__ Wx0t,   // [4H][V]  bf16 (transposed)
    const __hip_bfloat16* __restrict__ Wh0t,   // [4H][H]  bf16
    const float* __restrict__ b0,              // [4H]
    const __hip_bfloat16* __restrict__ Wx1t,   // [4H][H]  bf16
    const __hip_bfloat16* __restrict__ Wh1t,   // [4H][H]  bf16
    const float* __restrict__ b1,              // [4H]
    __hip_bfloat16* __restrict__ h0b0, __hip_bfloat16* __restrict__ h0b1,
    __hip_bfloat16* __restrict__ h1b0, __hip_bfloat16* __restrict__ h1b1,
    const float* __restrict__ h_in,            // [2][B][H]
    const float* __restrict__ c_in,            // [2][B][H]
    float* __restrict__ outs,                  // [B][T][H]
    float* __restrict__ hf,                    // [2][B][H] running/final h (fp32)
    float* __restrict__ cf,                    // [2][B][H] final c
    unsigned* __restrict__ bar)
{
    const int lane = threadIdx.x & 31;
    const int w    = (blockIdx.x * BLOCK_THREADS + threadIdx.x) >> 5;
    const int rt   = w >> 6;        // 0..7  batch tile
    const int ct   = w & 63;        // 0..63 hidden tile
    const int half = lane >> 4;
    const int nl   = lane & 15;
    const int row  = rt * 16 + nl;  // A-matrix row (batch) for this lane
    const int col  = ct * 16 + nl;  // N column within each gate

    __hip_bfloat16* h0b[2] = { h0b0, h0b1 };
    __hip_bfloat16* h1b[2] = { h1b0, h1b1 };

    // ---- seed: c state into registers, initial h into bf16 buffer 0 ----
    float c0r[8], c1r[8];
#pragma unroll
    for (int e = 0; e < 8; ++e) {
        const int rb = rt * 16 + e + half * 8;     // C/D: M = e + 8*(lane>=16)
        const size_t sidx = (size_t)rb * Hz + col;
        c0r[e] = c_in[sidx];  c1r[e] = c_in[BH + sidx];
        h0b0[sidx] = __float2bfloat16(h_in[sidx]);
        h1b0[sidx] = __float2bfloat16(h_in[BH + sidx]);
    }

    const float bi0 = b0[col],          bf0 = b0[Hz + col];
    const float bg0 = b0[2 * Hz + col], bo0 = b0[3 * Hz + col];
    const float bi1 = b1[col],          bf1 = b1[Hz + col];
    const float bg1 = b1[2 * Hz + col], bo1 = b1[3 * Hz + col];

    // per-lane 32-bit element offsets (shared across the three K=H matrices)
    int woffV[4], woffH[4];
#pragma unroll
    for (int g = 0; g < 4; ++g) {
        woffV[g] = (g * Hz + col) * Vz;
        woffH[g] = (g * Hz + col) * Hz;
    }
    const int aoffH = row * Hz;

    unsigned lgen = 0;
    grid_barrier(bar, &lgen);

    for (int t = 0; t < Tz; ++t) {
        const int cur = t & 1, nxt = cur ^ 1;

        // Opaque zero: makes weight-fragment addresses non-loop-invariant so
        // the compiler cannot hoist (and then spill) t-invariant B loads,
        // while keeping the pointers themselves in the global address space.
        int zo = 0;
        asm volatile("" : "+s"(zo));

        // ---------------- layer 0 ----------------
        {
            v8f acc[4] = {};
            wmma_kloop(acc, xbf, (row * Tz + t) * Vz, Wx0t, woffV, zo, Vz, half);
            wmma_kloop(acc, h0b[cur], aoffH,          Wh0t, woffH, zo, Hz, half);
#pragma unroll
            for (int e = 0; e < 8; ++e) {
                const int rb = rt * 16 + e + half * 8;
                const size_t sidx = (size_t)rb * Hz + col;
                const float iv = fast_sigmoid(acc[0][e] + bi0);
                const float fv = fast_sigmoid(acc[1][e] + bf0);
                const float gv = fast_tanh   (acc[2][e] + bg0);
                const float ov = fast_sigmoid(acc[3][e] + bo0);
                const float cn = fv * c0r[e] + iv * gv;
                const float hn = ov * fast_tanh(cn);
                c0r[e] = cn;
                h0b[nxt][sidx] = __float2bfloat16(hn);
                hf[sidx] = hn;                       // running fp32 h (h_f[0])
            }
        }
        grid_barrier(bar, &lgen);   // h0_new visible device-wide

        // ---------------- layer 1 ----------------
        {
            v8f acc[4] = {};
            wmma_kloop(acc, h0b[nxt], aoffH, Wx1t, woffH, zo, Hz, half);
            wmma_kloop(acc, h1b[cur], aoffH, Wh1t, woffH, zo, Hz, half);
#pragma unroll
            for (int e = 0; e < 8; ++e) {
                const int rb = rt * 16 + e + half * 8;
                const size_t sidx = (size_t)rb * Hz + col;
                const float iv = fast_sigmoid(acc[0][e] + bi1);
                const float fv = fast_sigmoid(acc[1][e] + bf1);
                const float gv = fast_tanh   (acc[2][e] + bg1);
                const float ov = fast_sigmoid(acc[3][e] + bo1);
                const float cn = fv * c1r[e] + iv * gv;
                const float hn = ov * fast_tanh(cn);
                c1r[e] = cn;
                h1b[nxt][sidx] = __float2bfloat16(hn);
                hf[BH + sidx] = hn;                  // running fp32 h (h_f[1])
                outs[((size_t)rb * Tz + t) * Hz + col] = hn;
            }
        }
        grid_barrier(bar, &lgen);   // h1_new visible + WAR safety for next t
    }

    // ---- epilogue: final c from registers ----
#pragma unroll
    for (int e = 0; e < 8; ++e) {
        const int rb = rt * 16 + e + half * 8;
        const size_t sidx = (size_t)rb * Hz + col;
        cf[sidx]      = c0r[e];
        cf[BH + sidx] = c1r[e];
    }
}

// in[K][N] fp32 (row-major) -> out[N][K] bf16 (transposed, contiguous K)
__global__ void transpose_cvt(const float* __restrict__ in,
                              __hip_bfloat16* __restrict__ out,
                              int K, int N)
{
    const size_t idx = (size_t)blockIdx.x * blockDim.x + threadIdx.x;
    if (idx >= (size_t)K * N) return;
    const int n = (int)(idx / K);
    const int k = (int)(idx % K);
    out[idx] = __float2bfloat16(in[(size_t)k * N + n]);
}

__global__ void cvt_bf16(const float* __restrict__ in,
                         __hip_bfloat16* __restrict__ out, size_t n)
{
    const size_t idx = (size_t)blockIdx.x * blockDim.x + threadIdx.x;
    if (idx < n) out[idx] = __float2bfloat16(in[idx]);
}

__global__ void zero_bar(unsigned* __restrict__ bar)
{
    if (threadIdx.x < 64) bar[threadIdx.x] = 0u;
}

extern "C" void kernel_launch(void* const* d_in, const int* in_sizes, int n_in,
                              void* d_out, int out_size, void* d_ws, size_t ws_size,
                              hipStream_t stream)
{
    (void)in_sizes; (void)n_in; (void)out_size; (void)ws_size;

    const float* x   = (const float*)d_in[0];
    const float* h0i = (const float*)d_in[1];
    const float* c0i = (const float*)d_in[2];
    const float* Wx0 = (const float*)d_in[3];
    const float* Wh0 = (const float*)d_in[4];
    const float* b0  = (const float*)d_in[5];
    const float* Wx1 = (const float*)d_in[6];
    const float* Wh1 = (const float*)d_in[7];
    const float* b1  = (const float*)d_in[8];

    // -------- workspace layout --------
    char* ws = (char*)d_ws;
    size_t off = 0;
    __hip_bfloat16* Wx0t = (__hip_bfloat16*)(ws + off); off += (size_t)H4 * Vz * 2;
    __hip_bfloat16* Wh0t = (__hip_bfloat16*)(ws + off); off += (size_t)H4 * Hz * 2;
    __hip_bfloat16* Wx1t = (__hip_bfloat16*)(ws + off); off += (size_t)H4 * Hz * 2;
    __hip_bfloat16* Wh1t = (__hip_bfloat16*)(ws + off); off += (size_t)H4 * Hz * 2;
    __hip_bfloat16* xbf  = (__hip_bfloat16*)(ws + off); off += (size_t)Bz * Tz * Vz * 2;
    __hip_bfloat16* h0bA = (__hip_bfloat16*)(ws + off); off += BH * 2;
    __hip_bfloat16* h0bB = (__hip_bfloat16*)(ws + off); off += BH * 2;
    __hip_bfloat16* h1bA = (__hip_bfloat16*)(ws + off); off += BH * 2;
    __hip_bfloat16* h1bB = (__hip_bfloat16*)(ws + off); off += BH * 2;
    unsigned*       bar  = (unsigned*)(ws + off);       off += 256;

    // -------- d_out layout: outs | h_f(2,B,H) | c_f(2,B,H) --------
    float* outs = (float*)d_out;
    float* hf   = outs + (size_t)Bz * Tz * Hz;
    float* cf   = hf + 2 * BH;

    // -------- prep: weight transpose+cvt, x cvt, barrier init --------
    {
        size_t n;
        n = (size_t)H4 * Vz;
        transpose_cvt<<<dim3((unsigned)((n + 255) / 256)), dim3(256), 0, stream>>>(Wx0, Wx0t, Vz, H4);
        n = (size_t)H4 * Hz;
        transpose_cvt<<<dim3((unsigned)((n + 255) / 256)), dim3(256), 0, stream>>>(Wh0, Wh0t, Hz, H4);
        transpose_cvt<<<dim3((unsigned)((n + 255) / 256)), dim3(256), 0, stream>>>(Wx1, Wx1t, Hz, H4);
        transpose_cvt<<<dim3((unsigned)((n + 255) / 256)), dim3(256), 0, stream>>>(Wh1, Wh1t, Hz, H4);
        n = (size_t)Bz * Tz * Vz;
        cvt_bf16<<<dim3((unsigned)((n + 255) / 256)), dim3(256), 0, stream>>>(x, xbf, n);
        zero_bar<<<dim3(1), dim3(64), 0, stream>>>(bar);
    }

    // -------- single persistent kernel runs the whole 512-step scan --------
    lstm_persistent<<<dim3(GRID_BLOCKS), dim3(BLOCK_THREADS), 0, stream>>>(
        xbf, Wx0t, Wh0t, b0, Wx1t, Wh1t, b1,
        h0bA, h0bB, h1bA, h1bB,
        h0i, c0i, outs, hf, cf, bar);
}
